// DecoderWithAttention_18528488915169
// MI455X (gfx1250) — matrix operand order repaired and checked by reading, not compiled, not run on hardware
//
#include <hip/hip_runtime.h>

// Problem dims
#define B_     32
#define P_     196
#define FEAT_  2048
#define ATT_   512
#define HID_   512
#define EMB_   512
#define VOCAB_ 10000
#define T_     25
#define INP_   2560   // EMB + FEAT
#define NCAT_  4608   // W_dec(512) + W_beta(2048) + W_hh(2048)

typedef __attribute__((ext_vector_type(16))) __bf16 v16bf;
typedef __attribute__((ext_vector_type(8)))  __bf16 v8bf;
typedef __attribute__((ext_vector_type(8)))  float  v8f;
typedef __attribute__((ext_vector_type(4)))  float  v4f;

__device__ __forceinline__ __bf16 f2bf(float f) {
  union { float f; unsigned u; } v; v.f = f;
  unsigned r = v.u + 0x7FFFu + ((v.u >> 16) & 1u);   // round-to-nearest-even
  unsigned short h = (unsigned short)(r >> 16);
  __bf16 b; __builtin_memcpy(&b, &h, 2); return b;
}
__device__ __forceinline__ float sigf(float x) { return 1.0f / (1.0f + __expf(-x)); }

// ---------------- WMMA fragment loaders (ISA 7.12.2 layouts, wave32) ------------
// A 16x32 bf16: lanes 0-15 row M=lane, V0..3 hold K=0..7, V4..7 hold K=16..23;
// lanes 16-31 same rows, K ranges +8. -> two 16B chunks at k = half*8 and 16+half*8.
__device__ __forceinline__ v16bf load_a_bf16(const __bf16* A, int lda, int m0, int k0, int lane) {
  int half = lane >> 4;
  const __bf16* row = A + (size_t)(m0 + (lane & 15)) * lda + k0 + half * 8;
  v8bf lo = *(const v8bf*)(row);
  v8bf hi = *(const v8bf*)(row + 16);
  v16bf a;
#pragma unroll
  for (int i = 0; i < 8; ++i) { a[i] = lo[i]; a[8 + i] = hi[i]; }
  return a;
}
// Same layout but source is fp32; convert in-register (enc is read once from HBM).
__device__ __forceinline__ v16bf load_a_f32(const float* A, int lda, int m0, int k0, int lane) {
  int half = lane >> 4;
  const float* row = A + (size_t)(m0 + (lane & 15)) * lda + k0 + half * 8;
  v4f a0 = *(const v4f*)(row),      a1 = *(const v4f*)(row + 4);
  v4f b0 = *(const v4f*)(row + 16), b1 = *(const v4f*)(row + 20);
  v16bf a;
#pragma unroll
  for (int i = 0; i < 4; ++i) {
    a[i]     = f2bf(a0[i]); a[4 + i]  = f2bf(a1[i]);
    a[8 + i] = f2bf(b0[i]); a[12 + i] = f2bf(b1[i]);
  }
  return a;
}
// B 32x16 bf16 (KxN), weights stored row-major [N,K] so B[k][n] = W[n][k]:
// lane n = lane&15, same K order as A: two 16B chunks at k = half*8 and 16+half*8.
__device__ __forceinline__ v16bf load_b(const __bf16* Bm, int ldb, int n0, int k0, int lane) {
  int half = lane >> 4;
  const __bf16* row = Bm + (size_t)(n0 + (lane & 15)) * ldb + k0 + half * 8;
  v8bf lo = *(const v8bf*)(row);
  v8bf hi = *(const v8bf*)(row + 16);
  v16bf b;
#pragma unroll
  for (int i = 0; i < 8; ++i) { b[i] = lo[i]; b[8 + i] = hi[i]; }
  return b;
}

__device__ __forceinline__ void store_tile(float* C, int ldc, const float* bias,
                                           const float* Cadd, int ldcadd,
                                           int m0, int n0, int lane, v8f acc) {
  int half = lane >> 4, col = lane & 15;
  int c = n0 + col;
  float bv = bias ? bias[c] : 0.0f;
#pragma unroll
  for (int r = 0; r < 8; ++r) {
    int row = m0 + r + half * 8;      // C layout: VGPR r -> M = r (+8 for upper half)
    float v = acc[r] + bv;
    if (Cadd) v += Cadd[(size_t)row * ldcadd + c];
    C[(size_t)row * (size_t)ldc + c] = v;
  }
}

#define WMMA_BF16(ACC, A, Bv) \
  ACC = __builtin_amdgcn_wmma_f32_16x16x32_bf16(false, A, false, Bv, (short)0, ACC, false, false)

// ---------------- Generic bf16 WMMA GEMM: C[M,N] = A[M,K] * B[N,K]^T (+bias)(+Cadd)
// One wave owns a 16-row M tile and 4 consecutive 16-col N tiles (A reuse x4).
// Wave/tile indices are pushed through readfirstlane so all tile guards are
// scalar branches: EXEC provably stays all-ones around every WMMA (ISA 7.12 req).
template <bool A_F32>
__global__ __launch_bounds__(256) void gemm_bf16(const void* __restrict__ Av, int lda,
                                                 const __bf16* __restrict__ Bm, int ldb,
                                                 const float* __restrict__ bias,
                                                 const float* __restrict__ Cadd, int ldcadd,
                                                 float* __restrict__ C, int ldc,
                                                 int M, int N, int K) {
  const int NT = 4;
  int wave = (blockIdx.x * (int)blockDim.x + (int)threadIdx.x) >> 5;
  wave = __builtin_amdgcn_readfirstlane(wave);          // wave-uniform -> SGPR
  int lane = threadIdx.x & 31;
  int mtiles = M >> 4, ntiles = N >> 4;
  int ngroups = (ntiles + NT - 1) / NT;
  if (wave >= mtiles * ngroups) return;                 // scalar branch
  int mt = wave / ngroups, ng = wave - mt * ngroups;
  int m0 = mt << 4;
  int ntb = ng * NT;
  int ncnt = ntiles - ntb; if (ncnt > NT) ncnt = NT;    // SGPR

  v8f acc0 = {}, acc1 = {}, acc2 = {}, acc3 = {};
  if (ncnt == NT) {
    // hot path: issue A + all 4 B loads first, then 4 back-to-back WMMAs
    for (int k0 = 0; k0 < K; k0 += 32) {
      v16bf a;
      if constexpr (A_F32) a = load_a_f32((const float*)Av, lda, m0, k0, lane);
      else                 a = load_a_bf16((const __bf16*)Av, lda, m0, k0, lane);
      v16bf b0 = load_b(Bm, ldb, (ntb + 0) << 4, k0, lane);
      v16bf b1 = load_b(Bm, ldb, (ntb + 1) << 4, k0, lane);
      v16bf b2 = load_b(Bm, ldb, (ntb + 2) << 4, k0, lane);
      v16bf b3 = load_b(Bm, ldb, (ntb + 3) << 4, k0, lane);
      if (k0 + 32 < K)  // pull next K slab of B toward cache (global_prefetch_b8)
        __builtin_prefetch((const char*)(Bm + (size_t)(ntb << 4) * ldb + k0 + 32), 0, 1);
      WMMA_BF16(acc0, a, b0);
      WMMA_BF16(acc1, a, b1);
      WMMA_BF16(acc2, a, b2);
      WMMA_BF16(acc3, a, b3);
    }
  } else {
    // ragged tail group (only the last group of W_fc's 625 N-tiles)
    for (int k0 = 0; k0 < K; k0 += 32) {
      v16bf a;
      if constexpr (A_F32) a = load_a_f32((const float*)Av, lda, m0, k0, lane);
      else                 a = load_a_bf16((const __bf16*)Av, lda, m0, k0, lane);
      {
        v16bf b = load_b(Bm, ldb, (ntb + 0) << 4, k0, lane);
        WMMA_BF16(acc0, a, b);
      }
      if (ncnt > 1) { v16bf b = load_b(Bm, ldb, (ntb + 1) << 4, k0, lane); WMMA_BF16(acc1, a, b); }
      if (ncnt > 2) { v16bf b = load_b(Bm, ldb, (ntb + 2) << 4, k0, lane); WMMA_BF16(acc2, a, b); }
    }
  }
  store_tile(C, ldc, bias, Cadd, ldcadd, m0, (ntb + 0) << 4, lane, acc0);
  if (ncnt > 1) store_tile(C, ldc, bias, Cadd, ldcadd, m0, (ntb + 1) << 4, lane, acc1);
  if (ncnt > 2) store_tile(C, ldc, bias, Cadd, ldcadd, m0, (ntb + 2) << 4, lane, acc2);
  if (ncnt > 3) store_tile(C, ldc, bias, Cadd, ldcadd, m0, (ntb + 3) << 4, lane, acc3);
}

// ---------------- helper kernels --------------------------------------------
__global__ void cvt_f32_bf16(const float* __restrict__ s, __bf16* __restrict__ d, int n) {
  int i = blockIdx.x * blockDim.x + threadIdx.x;
  if (i < n) d[i] = f2bf(s[i]);
}

// teacher-forcing gather: prewords = [<start>=1, captions[:, :-1]]
__global__ void emb_gather(const int* __restrict__ caps, const float* __restrict__ emb,
                           __bf16* __restrict__ out) {
  int bt = blockIdx.x;               // b*T + t, 800 blocks x 512 threads
  int b = bt / T_, t = bt - b * T_;
  int word = (t == 0) ? 1 : caps[b * T_ + (t - 1)];
  int e = threadIdx.x;
  out[(size_t)bt * EMB_ + e] = f2bf(emb[(size_t)word * EMB_ + e]);
}

__global__ void mean_enc_k(const float* __restrict__ enc, float* __restrict__ mean) {
  int i = blockIdx.x * blockDim.x + threadIdx.x;      // B*FEAT
  if (i >= B_ * FEAT_) return;
  int b = i / FEAT_, f = i - b * FEAT_;
  const float* p = enc + (size_t)b * P_ * FEAT_ + f;
  float s = 0.f;
  for (int q = 0; q < P_; ++q) s += p[(size_t)q * FEAT_];
  mean[i] = s * (1.0f / P_);
}

__global__ void init_hc(const float* __restrict__ mean,
                        const float* __restrict__ Wh, const float* __restrict__ bh,
                        const float* __restrict__ Wc, const float* __restrict__ bc,
                        __bf16* __restrict__ hbf, float* __restrict__ c) {
  int i = blockIdx.x * blockDim.x + threadIdx.x;      // 2*B*HID
  if (i >= 2 * B_ * HID_) return;
  int which = i / (B_ * HID_);
  int r = i - which * (B_ * HID_);
  int b = r / HID_, j = r - b * HID_;
  const float* W = (which ? Wc : Wh) + (size_t)j * FEAT_;
  const float* m = mean + (size_t)b * FEAT_;
  float s = which ? bc[j] : bh[j];
  for (int k = 0; k < FEAT_; ++k) s += m[k] * W[k];
  if (which) c[r] = s; else hbf[r] = f2bf(s);
}

// attention: scores = relu(att1 + att2) @ W_full; softmax; awe = alpha-weighted enc;
// gated awe + emb_t packed into z (bf16) for the W_ih GEMM. One block per batch row.
__global__ __launch_bounds__(256) void attention_k(const float* __restrict__ att1,
                                                   const float* __restrict__ catOut,
                                                   const float* __restrict__ enc,
                                                   const float* __restrict__ W_full,
                                                   const float* __restrict__ b_full,
                                                   const __bf16* __restrict__ embeds, int t,
                                                   float* __restrict__ alphas,
                                                   __bf16* __restrict__ z) {
  __shared__ float s_att2[ATT_];
  __shared__ float s_sc[P_];
  __shared__ float s_red[2];
  int b = blockIdx.x, tid = threadIdx.x;
  for (int i = tid; i < ATT_; i += 256) s_att2[i] = catOut[b * NCAT_ + i];
  __syncthreads();
  if (tid < P_) {
    const float* ar = att1 + ((size_t)b * P_ + tid) * ATT_;
    float s = 0.f;
#pragma unroll 4
    for (int a = 0; a < ATT_; ++a) {
      float v = ar[a] + s_att2[a];
      s += (v > 0.f ? v : 0.f) * W_full[a];
    }
    s_sc[tid] = s + b_full[0];
  }
  __syncthreads();
  if (tid == 0) { float m = -3.0e38f; for (int p = 0; p < P_; ++p) m = fmaxf(m, s_sc[p]); s_red[0] = m; }
  __syncthreads();
  if (tid < P_) s_sc[tid] = __expf(s_sc[tid] - s_red[0]);
  __syncthreads();
  if (tid == 0) { float s = 0.f; for (int p = 0; p < P_; ++p) s += s_sc[p]; s_red[1] = 1.f / s; }
  __syncthreads();
  if (tid < P_) {
    float al = s_sc[tid] * s_red[1];
    s_sc[tid] = al;
    alphas[((size_t)b * T_ + t) * P_ + tid] = al;
  }
  __syncthreads();
  for (int f = tid; f < FEAT_; f += 256) {
    float s = 0.f;
    const float* ep = enc + (size_t)b * P_ * FEAT_ + f;
    for (int p = 0; p < P_; ++p) s += s_sc[p] * ep[(size_t)p * FEAT_];
    float g = sigf(catOut[b * NCAT_ + ATT_ + f]);
    z[(size_t)b * INP_ + EMB_ + f] = f2bf(g * s);
  }
  for (int e = tid; e < EMB_; e += 256)
    z[(size_t)b * INP_ + e] = embeds[((size_t)b * T_ + t) * EMB_ + e];
}

__global__ __launch_bounds__(256) void lstm_cell_k(const float* __restrict__ gates,
                                                   float* __restrict__ c,
                                                   __bf16* __restrict__ hbf) {
  int i = blockIdx.x * 256 + threadIdx.x;             // B*HID
  if (i >= B_ * HID_) return;
  int b = i >> 9, j = i & (HID_ - 1);
  const float* g = gates + (size_t)b * 4 * HID_;
  float ig = sigf(g[j]);
  float fg = sigf(g[HID_ + j]);
  float gg = tanhf(g[2 * HID_ + j]);
  float og = sigf(g[3 * HID_ + j]);
  float cn = fg * c[i] + ig * gg;
  c[i] = cn;
  hbf[i] = f2bf(og * tanhf(cn));
}

// ---------------- host side ---------------------------------------------------
template <bool AF>
static void launch_gemm(const void* A, int lda, const __bf16* Bm, int ldb,
                        const float* bias, const float* Cadd, int ldcadd,
                        float* C, int ldc, int M, int N, int K, hipStream_t s) {
  int mt = M >> 4, nt = N >> 4, ng = (nt + 3) / 4;
  int waves = mt * ng;
  int blocks = (waves + 7) / 8;                        // 8 waves / 256-thread block
  gemm_bf16<AF><<<blocks, 256, 0, s>>>(A, lda, Bm, ldb, bias, Cadd, ldcadd, C, ldc, M, N, K);
}

extern "C" void kernel_launch(void* const* d_in, const int* in_sizes, int n_in,
                              void* d_out, int out_size, void* d_ws, size_t ws_size,
                              hipStream_t stream) {
  const float* enc      = (const float*)d_in[0];
  const int*   caps     = (const int*)d_in[1];
  const float* W_enc    = (const float*)d_in[2];
  const float* b_enc    = (const float*)d_in[3];
  const float* W_dec    = (const float*)d_in[4];
  const float* b_dec    = (const float*)d_in[5];
  const float* W_full   = (const float*)d_in[6];
  const float* b_full   = (const float*)d_in[7];
  const float* emb      = (const float*)d_in[8];
  const float* W_ih     = (const float*)d_in[9];
  const float* b_ih     = (const float*)d_in[10];
  const float* W_hh     = (const float*)d_in[11];
  const float* b_hh     = (const float*)d_in[12];
  const float* W_init_h = (const float*)d_in[13];
  const float* b_init_h = (const float*)d_in[14];
  const float* W_init_c = (const float*)d_in[15];
  const float* b_init_c = (const float*)d_in[16];
  const float* W_beta   = (const float*)d_in[17];
  const float* b_beta   = (const float*)d_in[18];
  const float* W_fc     = (const float*)d_in[19];
  const float* b_fc     = (const float*)d_in[20];
  (void)in_sizes; (void)n_in; (void)out_size; (void)ws_size;

  char* ws = (char*)d_ws;
  size_t off = 0;
  auto alloc = [&](size_t bytes) -> char* {
    char* p = ws + off;
    off = (off + bytes + 255) & ~(size_t)255;
    return p;
  };
  __bf16* Wcat   = (__bf16*)alloc((size_t)NCAT_ * HID_ * 2);   // [Wdec;Wbeta;Whh] bf16 [4608,512]
  __bf16* Wihb   = (__bf16*)alloc((size_t)4 * HID_ * INP_ * 2);
  __bf16* Wfcb   = (__bf16*)alloc((size_t)VOCAB_ * HID_ * 2);
  __bf16* Wencb  = (__bf16*)alloc((size_t)ATT_ * FEAT_ * 2);
  float*  bcat   = (float*)alloc((size_t)NCAT_ * 4);
  float*  att1   = (float*)alloc((size_t)B_ * P_ * ATT_ * 4);
  __bf16* embeds = (__bf16*)alloc((size_t)B_ * T_ * EMB_ * 2);
  float*  meanE  = (float*)alloc((size_t)B_ * FEAT_ * 4);
  float*  catOut = (float*)alloc((size_t)B_ * NCAT_ * 4);      // [att2 | gatePre | hWhh]
  __bf16* zbuf   = (__bf16*)alloc((size_t)B_ * INP_ * 2);      // [emb_t | gated awe]
  float*  gates  = (float*)alloc((size_t)B_ * 4 * HID_ * 4);
  float*  cst    = (float*)alloc((size_t)B_ * HID_ * 4);
  __bf16* hbf    = (__bf16*)alloc((size_t)B_ * HID_ * 2);

  float* preds  = (float*)d_out;                               // [32,25,10000]
  float* alphas = preds + (size_t)B_ * T_ * VOCAB_;            // [32,25,196]

  // --- one-time: bf16 weight conversion + bias concat + gathers + att1 ---
  auto cvt = [&](const float* s, __bf16* d, int n) {
    cvt_f32_bf16<<<(n + 255) / 256, 256, 0, stream>>>(s, d, n);
  };
  cvt(W_dec,  Wcat,                       ATT_ * HID_);
  cvt(W_beta, Wcat + (size_t)ATT_ * HID_, FEAT_ * HID_);
  cvt(W_hh,   Wcat + (size_t)(ATT_ + FEAT_) * HID_, 4 * HID_ * HID_);
  cvt(W_ih,   Wihb,  4 * HID_ * INP_);
  cvt(W_fc,   Wfcb,  VOCAB_ * HID_);
  cvt(W_enc,  Wencb, ATT_ * FEAT_);
  hipMemcpyAsync(bcat,                b_dec,  ATT_  * 4, hipMemcpyDeviceToDevice, stream);
  hipMemcpyAsync(bcat + ATT_,         b_beta, FEAT_ * 4, hipMemcpyDeviceToDevice, stream);
  hipMemcpyAsync(bcat + ATT_ + FEAT_, b_hh,   4 * HID_ * 4, hipMemcpyDeviceToDevice, stream);
  emb_gather<<<B_ * T_, EMB_, 0, stream>>>(caps, emb, embeds);
  mean_enc_k<<<(B_ * FEAT_ + 255) / 256, 256, 0, stream>>>(enc, meanE);
  init_hc<<<(2 * B_ * HID_ + 255) / 256, 256, 0, stream>>>(meanE, W_init_h, b_init_h,
                                                           W_init_c, b_init_c, hbf, cst);
  // att1[6272,512] = enc(f32->bf16 in-register) @ W_enc^T + b_enc
  launch_gemm<true>(enc, FEAT_, Wencb, FEAT_, b_enc, nullptr, 0,
                    att1, ATT_, B_ * P_, ATT_, FEAT_, stream);

  // --- recurrent steps ---
  for (int t = 0; t < T_; ++t) {
    // catOut = h @ [W_dec;W_beta;W_hh]^T + [b_dec;b_beta;b_hh]   (M=32,N=4608,K=512)
    launch_gemm<false>(hbf, HID_, Wcat, HID_, bcat, nullptr, 0,
                       catOut, NCAT_, B_, NCAT_, HID_, stream);
    attention_k<<<B_, 256, 0, stream>>>(att1, catOut, enc, W_full, b_full,
                                        embeds, t, alphas, zbuf);
    // gates = z @ W_ih^T + b_ih + (h @ W_hh^T + b_hh)            (M=32,N=2048,K=2560)
    launch_gemm<false>(zbuf, INP_, Wihb, INP_, b_ih,
                       catOut + (ATT_ + FEAT_), NCAT_,
                       gates, 4 * HID_, B_, 4 * HID_, INP_, stream);
    lstm_cell_k<<<(B_ * HID_ + 255) / 256, 256, 0, stream>>>(gates, cst, hbf);
    // preds[:,t,:] = h_new @ W_fc^T + b_fc                        (M=32,N=10000,K=512)
    launch_gemm<false>(hbf, HID_, Wfcb, HID_, b_fc, nullptr, 0,
                       preds + (size_t)t * VOCAB_, T_ * VOCAB_, B_, VOCAB_, HID_, stream);
  }
}
